// SinkhornAttention_22265110463115
// MI455X (gfx1250) — matrix operand order repaired and checked by reading, not compiled, not run on hardware
//
#include <hip/hip_runtime.h>
#include <hip/hip_bf16.h>
#include <math.h>

typedef __bf16 v16bf __attribute__((ext_vector_type(16)));
typedef float  v8f   __attribute__((ext_vector_type(8)));
typedef unsigned int u32x4 __attribute__((ext_vector_type(4)));
typedef int          i32x4 __attribute__((ext_vector_type(4)));
typedef int          i32x8 __attribute__((ext_vector_type(8)));

#define TT    4096
#define DH    64
#define BS    256
#define NB    16
#define BHH   32
#define SCALE 0.125f
#define EPSV  1e-6f
#define TEMP  0.75f
#define SINK_ITERS 7

// ---------------------------------------------------------------------------
// WMMA fragment loaders (CDNA5 ISA 7.12.2 layouts, wave32)
// ---------------------------------------------------------------------------
__device__ inline v16bf load_a_frag(const __bf16* base, int ld, int m0, int k0, int lane) {
    int row = m0 + (lane & 15);
    int c0  = k0 + ((lane >> 4) << 3);
    const __bf16* p = base + row * ld + c0;
    v16bf a;
#pragma unroll
    for (int e = 0; e < 8; ++e) { a[e] = p[e]; a[8 + e] = p[16 + e]; }
    return a;
}

__device__ inline v16bf load_b_frag(const __bf16* base, int ld, int n0, int k0, int lane) {
    int row = n0 + (lane & 15);
    int c0  = k0 + ((lane >> 4) << 3);
    const __bf16* p = base + row * ld + c0;
    v16bf b;
#pragma unroll
    for (int e = 0; e < 8; ++e) { b[e] = p[e]; b[8 + e] = p[16 + e]; }
    return b;
}

// ---------------------------------------------------------------------------
// TDM: DMA a 2D tile [rows x cols] of fp32 (row-major, stride==cols) from
// global memory into LDS at byte offset lds_off. D# per ISA 08 §8.
// ---------------------------------------------------------------------------
__device__ inline void tdm_load_2d_f32(unsigned lds_off, const void* gaddr,
                                       unsigned rows, unsigned cols) {
    unsigned long long ga = (unsigned long long)gaddr;
    u32x4 g0;
    g0[0] = 1u;                                           // count=1 (valid), user mode
    g0[1] = lds_off;                                      // lds_addr [63:32]
    g0[2] = (unsigned)(ga & 0xFFFFFFFFu);                 // global_addr lo
    g0[3] = (unsigned)((ga >> 32) & 0x01FFFFFFu)          // global_addr [56:32]
          | (2u << 30);                                   // type=2 ("image")
    i32x8 g1;
    g1[0] = (int)(2u << 16);          // wg_mask=0, data_size=2 (4B), no pad/iter
    g1[1] = (int)(cols << 16);        // tensor_dim0[15:0]  -> bits [79:64... (48+16)]
    g1[2] = (int)(rows << 16);        // tensor_dim0 hi=0 | tensor_dim1[15:0]
    g1[3] = (int)(cols << 16);        // tensor_dim1 hi=0 | tile_dim0
    g1[4] = (int)rows;                // tile_dim1 | tile_dim2=0
    g1[5] = (int)cols;                // tensor_dim0_stride lo (48b)
    g1[6] = 0;                        // stride hi | tensor_dim1_stride lo
    g1[7] = 0;
    i32x4 gz = {0, 0, 0, 0};
#if defined(__clang_major__) && (__clang_major__ >= 23)
    i32x8 gz8 = {0, 0, 0, 0, 0, 0, 0, 0};
    __builtin_amdgcn_tensor_load_to_lds(g0, g1, gz, gz, gz8, 0);
#else
    __builtin_amdgcn_tensor_load_to_lds(g0, g1, gz, gz, 0);
#endif
}

// ---------------------------------------------------------------------------
// Kernel 1: per-bucket q/k sums  [bh*nb blocks, 64 threads]
// ---------------------------------------------------------------------------
__global__ void sk_bucket_sums(const float* __restrict__ q, const float* __restrict__ k,
                               float* __restrict__ qsum, float* __restrict__ ksum) {
    int blk = blockIdx.x;            // bh*NB + n
    int d   = threadIdx.x;           // 0..63
    size_t base = (size_t)blk * BS * DH;
    float sq = 0.f, sk = 0.f;
    for (int s = 0; s < BS; ++s) {
        sq += q[base + (size_t)s * DH + d];
        sk += k[base + (size_t)s * DH + d];
    }
    qsum[(size_t)blk * DH + d] = sq;
    ksum[(size_t)blk * DH + d] = sk;
}

// ---------------------------------------------------------------------------
// Kernel 2: gumbel-sinkhorn soft permutation  [bh blocks, 256 threads]
// ---------------------------------------------------------------------------
__global__ void sk_sinkhorn(const float* __restrict__ qsum, const float* __restrict__ ksum,
                            const float* __restrict__ gu, float* __restrict__ Rm) {
    __shared__ float M[16][16];
    int bh  = blockIdx.x;
    int tid = threadIdx.x;
    int i = tid >> 4, j = tid & 15;
    const float* qv = qsum + ((size_t)bh * NB + i) * DH;
    const float* kv = ksum + ((size_t)bh * NB + j) * DH;
    float dot = 0.f;
#pragma unroll 8
    for (int d = 0; d < DH; ++d) dot += qv[d] * kv[d];
    float r0 = fmaxf(dot * SCALE, 0.f);
    float u  = gu[(size_t)bh * 256 + tid];
    float g  = -__logf(-__logf(u + EPSV) + EPSV);
    float r  = (__logf(r0 + EPSV) + g) / TEMP;
    for (int it = 0; it < SINK_ITERS; ++it) {
        M[i][j] = r; __syncthreads();
        float mx = -1e30f;
        for (int t = 0; t < 16; ++t) mx = fmaxf(mx, M[i][t]);
        float s = 0.f;
        for (int t = 0; t < 16; ++t) s += __expf(M[i][t] - mx);
        r -= mx + __logf(s);
        __syncthreads();
        M[i][j] = r; __syncthreads();
        mx = -1e30f;
        for (int t = 0; t < 16; ++t) mx = fmaxf(mx, M[t][j]);
        s = 0.f;
        for (int t = 0; t < 16; ++t) s += __expf(M[t][j] - mx);
        r -= mx + __logf(s);
        __syncthreads();
    }
    Rm[(size_t)bh * 256 + tid] = __expf(r);
}

// ---------------------------------------------------------------------------
// LDS layout constants (bytes) for the attention kernel
// ---------------------------------------------------------------------------
#define OFF_QS    0
#define OFF_KL    (OFF_QS  + BS * DH * 2)
#define OFF_VTL   (OFF_KL  + BS * DH * 2)
#define OFF_KSR   (OFF_VTL + DH * BS * 2)
#define OFF_VTS   (OFF_KSR + BS * DH * 2)
#define OFF_WT    (OFF_VTS + DH * BS * 2)
#define OFF_SB    (OFF_WT  + DH * 128 * 2)      // 8 waves x [16][256] f16 (64KB)
#define OFF_PST   (OFF_SB  + 8 * 16 * 256 * 2)  // 8 waves x [16][32] bf16
#define OFF_OBUF  (OFF_PST + 8 * 16 * 32 * 2)   // [256][128] bf16 (64KB)
#define OFF_RROW  (OFF_OBUF + 256 * 128 * 2)
#define SMEM_BYTES ((size_t)OFF_RROW + 16 * 4)

// ---------------------------------------------------------------------------
// Kernel 3: fused local+sorted attention + output projection
// one block per (bh, bucket); 256 threads = 8 wave32
// K/V local tiles are DMAed by the TDM into the (initially dead) Sb/Obuf
// regions while the VALU-heavy soft-sort mixing loop runs.
// ---------------------------------------------------------------------------
__global__ void sk_attention(const float* __restrict__ q, const float* __restrict__ k,
                             const float* __restrict__ v, const float* __restrict__ Rm,
                             const float* __restrict__ W, const float* __restrict__ bo,
                             float* __restrict__ out) {
    extern __shared__ char smem_raw[];
    __bf16*   Qs  = (__bf16*)(smem_raw + OFF_QS);    // [256][64]
    __bf16*   Kl  = (__bf16*)(smem_raw + OFF_KL);    // [256][64]
    __bf16*   Vtl = (__bf16*)(smem_raw + OFF_VTL);   // [64][256]
    __bf16*   Ksr = (__bf16*)(smem_raw + OFF_KSR);   // [256][64]
    __bf16*   Vts = (__bf16*)(smem_raw + OFF_VTS);   // [64][256]
    __bf16*   Wt  = (__bf16*)(smem_raw + OFF_WT);    // [64][128]
    _Float16* Sb  = (_Float16*)(smem_raw + OFF_SB);  // logits staging
    __bf16*   Pst = (__bf16*)(smem_raw + OFF_PST);   // P staging
    __bf16*   Obuf= (__bf16*)(smem_raw + OFF_OBUF);  // concat(local,sorted)
    float*    Rrow= (float*)(smem_raw + OFF_RROW);   // [16]
    float*    KstageF = (float*)(smem_raw + OFF_SB);   // TDM dst: K fp32 tile
    float*    VstageF = (float*)(smem_raw + OFF_OBUF); // TDM dst: V fp32 tile

    int tid = threadIdx.x;
    int bh  = blockIdx.x >> 4;
    int n   = blockIdx.x & 15;
    size_t bbase = (size_t)bh * TT * DH;
    size_t nbase = bbase + (size_t)n * BS * DH;
    int wave = tid >> 5, lane = tid & 31;

    // ---- kick off TDM DMA of the local K and V fp32 tiles into LDS staging ----
    if (wave == 0) {
        tdm_load_2d_f32(OFF_SB,   k + nbase, BS, DH);
        tdm_load_2d_f32(OFF_OBUF, v + nbase, BS, DH);
    }
    if (tid < 16) Rrow[tid] = Rm[(size_t)bh * 256 + n * 16 + tid];
    __syncthreads();

    // ---- Q load + on-the-fly soft bucket mixing (reads hit L2), overlaps DMA ----
    for (int pos = tid; pos < BS * DH; pos += 256) {
        int s = pos >> 6, d = pos & 63;
        __builtin_prefetch(&q[nbase + pos + 4096], 0, 1);
        Qs[pos] = (__bf16)q[nbase + pos];
        float ak = 0.f, av = 0.f;
#pragma unroll
        for (int j = 0; j < NB; ++j) {
            float  rj  = Rrow[j];
            size_t off = bbase + (size_t)j * BS * DH + pos;
            ak += rj * k[off];
            av += rj * v[off];
        }
        Ksr[pos] = (__bf16)ak;
        Vts[d * BS + s] = (__bf16)av;
    }
    for (int pos = tid; pos < 128 * 64; pos += 256) {
        int kk = pos >> 6, nn = pos & 63;            // W [128][64] -> Wt [64][128]
        Wt[nn * 128 + kk] = (__bf16)W[pos];
    }

    // ---- drain the tensor DMA, then convert staged fp32 tiles to bf16 ----
    if (wave == 0) __builtin_amdgcn_s_wait_tensorcnt(0);
    __syncthreads();
    for (int pos = tid; pos < BS * DH; pos += 256) {
        int s = pos >> 6, d = pos & 63;
        Kl[pos]         = (__bf16)KstageF[pos];
        Vtl[d * BS + s] = (__bf16)VstageF[pos];
    }
    __syncthreads();

    int cl = lane & 15, hf = (lane >> 4) << 3;       // C layout: VGPR r <-> row r+hf
    int variant = wave >> 2;                         // 0=local, 1=sorted
    const __bf16* Ksrc = variant ? Ksr : Kl;
    const __bf16* Vsrc = variant ? Vts : Vtl;
    _Float16* Sw = Sb  + wave * 16 * 256;
    __bf16*   Pw = Pst + wave * 16 * 32;
    int wq = wave & 3;

    for (int qt = wq * 4; qt < wq * 4 + 4; ++qt) {
        v16bf aq0 = load_a_frag(Qs, 64, qt * 16, 0,  lane);
        v16bf aq1 = load_a_frag(Qs, 64, qt * 16, 32, lane);
        float mx[8];
#pragma unroll
        for (int r = 0; r < 8; ++r) mx[r] = -1e30f;

        // ---- phase A: S = Q K^T * scale, staged to LDS, track row max ----
        for (int nt = 0; nt < 16; ++nt) {
            v16bf b0 = load_b_frag(Ksrc, 64, nt * 16, 0,  lane);
            v16bf b1 = load_b_frag(Ksrc, 64, nt * 16, 32, lane);
            v8f s = {};
            s = __builtin_amdgcn_wmma_f32_16x16x32_bf16(false, aq0, false, b0, (short)0, s, false, false);
            s = __builtin_amdgcn_wmma_f32_16x16x32_bf16(false, aq1, false, b1, (short)0, s, false, false);
#pragma unroll
            for (int r = 0; r < 8; ++r) {
                float sv = s[r] * SCALE;
                mx[r] = fmaxf(mx[r], sv);
                Sw[(r + hf) * 256 + nt * 16 + cl] = (_Float16)sv;
            }
        }
#pragma unroll
        for (int r = 0; r < 8; ++r)
            for (int m = 1; m < 16; m <<= 1)
                mx[r] = fmaxf(mx[r], __shfl_xor(mx[r], m, 32));
        asm volatile("s_wait_dscnt 0" ::: "memory");

        // ---- phase B: P = exp(S - m), O += P V  (32 keys / iteration) ----
        v8f   oc[4] = {};
        float sum[8] = {};
        for (int nt2 = 0; nt2 < 8; ++nt2) {
#pragma unroll
            for (int t = 0; t < 2; ++t)
#pragma unroll
                for (int r = 0; r < 8; ++r) {
                    float sv = (float)Sw[(r + hf) * 256 + nt2 * 32 + t * 16 + cl];
                    float p  = __expf(sv - mx[r]);
                    sum[r] += p;
                    Pw[(r + hf) * 32 + t * 16 + cl] = (__bf16)p;   // C->A layout via LDS
                }
            asm volatile("s_wait_dscnt 0" ::: "memory");
            v16bf ap = load_a_frag(Pw, 32, 0, 0, lane);
#pragma unroll
            for (int dc = 0; dc < 4; ++dc) {
                v16bf bv = load_b_frag(Vsrc, 256, dc * 16, nt2 * 32, lane);
                oc[dc] = __builtin_amdgcn_wmma_f32_16x16x32_bf16(false, ap, false, bv, (short)0, oc[dc], false, false);
            }
        }
#pragma unroll
        for (int r = 0; r < 8; ++r) {
            for (int m = 1; m < 16; m <<= 1) sum[r] += __shfl_xor(sum[r], m, 32);
            sum[r] = 1.f / sum[r];
        }
#pragma unroll
        for (int dc = 0; dc < 4; ++dc)
#pragma unroll
            for (int r = 0; r < 8; ++r)
                Obuf[(qt * 16 + r + hf) * 128 + variant * 64 + dc * 16 + cl] =
                    (__bf16)(oc[dc][r] * sum[r]);
    }
    __syncthreads();

    // ---- output projection: [256,128] x Wt^T[128,64] + b_out ----
    for (int rti = 0; rti < 2; ++rti) {
        int rt = wave * 2 + rti;
#pragma unroll
        for (int cc = 0; cc < 4; ++cc) {
            v8f acc = {};
#pragma unroll
            for (int kc = 0; kc < 4; ++kc) {
                v16bf ao = load_a_frag(Obuf, 128, rt * 16, kc * 32, lane);
                v16bf bw = load_b_frag(Wt, 128, cc * 16, kc * 32, lane);
                acc = __builtin_amdgcn_wmma_f32_16x16x32_bf16(false, ao, false, bw, (short)0, acc, false, false);
            }
            float bias = bo[cc * 16 + cl];
#pragma unroll
            for (int r = 0; r < 8; ++r) {
                size_t row = (size_t)bh * TT + (size_t)n * BS + rt * 16 + r + hf;
                out[row * DH + cc * 16 + cl] = acc[r] + bias;
            }
        }
    }
}

// ---------------------------------------------------------------------------
extern "C" void kernel_launch(void* const* d_in, const int* in_sizes, int n_in,
                              void* d_out, int out_size, void* d_ws, size_t ws_size,
                              hipStream_t stream) {
    (void)in_sizes; (void)n_in; (void)out_size; (void)ws_size;
    const float* q  = (const float*)d_in[0];
    const float* k  = (const float*)d_in[1];
    const float* v  = (const float*)d_in[2];
    const float* gu = (const float*)d_in[3];
    const float* W  = (const float*)d_in[4];
    const float* bo = (const float*)d_in[5];
    float* out = (float*)d_out;

    float* ws   = (float*)d_ws;
    float* qsum = ws;                                    // [32*16*64]
    float* ksum = qsum + (size_t)BHH * NB * DH;          // [32*16*64]
    float* Rm   = ksum + (size_t)BHH * NB * DH;          // [32*16*16]

    sk_bucket_sums<<<dim3(BHH * NB), dim3(DH), 0, stream>>>(q, k, qsum, ksum);
    sk_sinkhorn<<<dim3(BHH), dim3(256), 0, stream>>>(qsum, ksum, gu, Rm);

    hipFuncSetAttribute((const void*)sk_attention,
                        hipFuncAttributeMaxDynamicSharedMemorySize, (int)SMEM_BYTES);
    sk_attention<<<dim3(BHH * NB), dim3(256), SMEM_BYTES, stream>>>(q, k, v, Rm, W, bo, out);
}
